// EagleModelABC_80848464380476
// MI455X (gfx1250) — compile-verified, matrix operand order checked by reading, not run on hardware
//
#include <hip/hip_runtime.h>
#include <math.h>

#define S_LEN 1024
#define H_DIM 4096
#define V_DIM 32000
#define TOPK 10
#define DEPTH 5
#define TOTAL 59
#define MPAD 16

typedef __attribute__((ext_vector_type(16))) __bf16 v16bf;
typedef __attribute__((ext_vector_type(8)))  __bf16 v8bf;
typedef __attribute__((ext_vector_type(8)))  float  v8f;

// ---------------------------------------------------------------------------
// B fragment from fp32 weight row: p = Wrow + k0 + koff (16B aligned).
// elements 0..7  <- fp32[p+0..7], elements 8..15 <- fp32[p+16..23]
// ---------------------------------------------------------------------------
__device__ __forceinline__ v16bf make_frag_f32(const float* p) {
    float4 a = *reinterpret_cast<const float4*>(p);
    float4 b = *reinterpret_cast<const float4*>(p + 4);
    float4 c = *reinterpret_cast<const float4*>(p + 16);
    float4 d = *reinterpret_cast<const float4*>(p + 20);
    v16bf f;
    f[0]  = (__bf16)a.x; f[1]  = (__bf16)a.y; f[2]  = (__bf16)a.z; f[3]  = (__bf16)a.w;
    f[4]  = (__bf16)b.x; f[5]  = (__bf16)b.y; f[6]  = (__bf16)b.z; f[7]  = (__bf16)b.w;
    f[8]  = (__bf16)c.x; f[9]  = (__bf16)c.y; f[10] = (__bf16)c.z; f[11] = (__bf16)c.w;
    f[12] = (__bf16)d.x; f[13] = (__bf16)d.y; f[14] = (__bf16)d.z; f[15] = (__bf16)d.w;
    return f;
}

// A fragment from pre-converted bf16 activations: two 16B loads, no cvt.
__device__ __forceinline__ v16bf make_frag_bf16(const __bf16* p) {
    v8bf lo = *reinterpret_cast<const v8bf*>(p);
    v8bf hi = *reinterpret_cast<const v8bf*>(p + 16);
    return __builtin_shufflevector(lo, hi, 0,1,2,3,4,5,6,7,8,9,10,11,12,13,14,15);
}

// ---------------------------------------------------------------------------
// Skinny GEMM: C[m,n] = sum_k A[m,k] * W[n,k] (+bias)(tanh), M=16 padded.
// A: bf16, row-major 16 x K. W: fp32, row-major N x K, streamed once.
// One wave per 16-column tile. mode: 0 none, 1 +bias, 2 tanh(x+bias).
// Writes fp32 C (optional) and bf16 Cbf (optional, next GEMM's A operand).
// ---------------------------------------------------------------------------
__global__ void k_gemm(const __bf16* __restrict__ A, const float* __restrict__ W,
                       const float* __restrict__ bias,
                       float* __restrict__ C, __bf16* __restrict__ Cbf,
                       int N, int K, long ldc, long ldcb, int mode, int store_rows) {
    int wave   = (blockIdx.x * blockDim.x + threadIdx.x) >> 5;
    int n_base = wave * 16;
    if (n_base >= N) return;
    int lane = threadIdx.x & 31;
    int hf   = lane >> 4;
    int l16  = lane & 15;
    int koff = hf * 8;
    const __bf16* Arow = A + (size_t)l16 * K + koff;
    int bcol = n_base + l16;
    const float* Wrow = W + (size_t)bcol * K + koff;

    v8f acc = {};
#pragma unroll 4
    for (int k0 = 0; k0 < K; k0 += 32) {
        __builtin_prefetch(Wrow + k0 + 512, 0, 0);   // global_prefetch_b8, stream ahead
        v16bf af = make_frag_bf16(Arow + k0);
        v16bf bf = make_frag_f32(Wrow + k0);
        acc = __builtin_amdgcn_wmma_f32_16x16x32_bf16(
                  false, af, false, bf, (short)0, acc, false, false);
    }
    float badd = (mode >= 1) ? bias[bcol] : 0.f;
    for (int j = 0; j < 8; ++j) {
        int row = j + hf * 8;
        float v = acc[j] + badd;
        if (mode == 2) v = tanhf(v);
        if (row < store_rows) {
            if (C)   C[(size_t)row * ldc + bcol] = v;
            if (Cbf) Cbf[(size_t)row * ldcb + bcol] = (__bf16)v;
        }
    }
}

// ---------------------------------------------------------------------------
// Small helpers (activations kept in bf16, concat layout [emb | hid], ld=2H)
// ---------------------------------------------------------------------------
__global__ void k_embed(const int* __restrict__ ids, const float* __restrict__ W_embed,
                        __bf16* __restrict__ abuf, int nrows) {
    int r = blockIdx.y;
    int c = blockIdx.x * blockDim.x + threadIdx.x;
    float v = (r < nrows) ? W_embed[(size_t)ids[r] * H_DIM + c] : 0.f;
    abuf[(size_t)r * (2 * H_DIM) + c] = (__bf16)v;
}

// mode 0: broadcast src row; mode 1: gather src[out_ids[r]] -> hid half of abuf
__global__ void k_set_hidden(const float* __restrict__ src, const int* __restrict__ out_ids,
                             __bf16* __restrict__ abuf, int mode, int nrows) {
    int r = blockIdx.y;
    int c = blockIdx.x * blockDim.x + threadIdx.x;
    float v = 0.f;
    if (r < nrows)
        v = (mode == 0) ? src[c] : src[(size_t)out_ids[r] * H_DIM + c];
    abuf[(size_t)r * (2 * H_DIM) + H_DIM + c] = (__bf16)v;
}

// per-row logsumexp over V (one block per row)
__global__ void k_row_reduce(const float* __restrict__ logits, float* __restrict__ row_lse) {
    int r = blockIdx.x, tid = threadIdx.x;
    __shared__ float red[256];
    float mx = -INFINITY;
    for (int c = tid; c < V_DIM; c += 256) mx = fmaxf(mx, logits[(size_t)r * V_DIM + c]);
    red[tid] = mx; __syncthreads();
    for (int s = 128; s > 0; s >>= 1) { if (tid < s) red[tid] = fmaxf(red[tid], red[tid + s]); __syncthreads(); }
    float m = red[0]; __syncthreads();
    float sum = 0.f;
    for (int c = tid; c < V_DIM; c += 256) sum += __expf(logits[(size_t)r * V_DIM + c] - m);
    red[tid] = sum; __syncthreads();
    for (int s = 128; s > 0; s >>= 1) { if (tid < s) red[tid] += red[tid + s]; __syncthreads(); }
    if (tid == 0) row_lse[r] = m + __logf(red[0]);
}

// per-row top-10 of logits (log_softmax monotone per row -> shift by lse)
__global__ void k_topk_row(const float* __restrict__ logits, const float* __restrict__ row_lse,
                           float* __restrict__ topk_p, int* __restrict__ topk_i) {
    int r = blockIdx.x, tid = threadIdx.x;
    __shared__ float rv[256];
    __shared__ int   ri[256];
    __shared__ int   sel[TOPK];
    for (int t = 0; t < TOPK; ++t) {
        float bv = -INFINITY; int bi = 0x7fffffff;
        for (int c = tid; c < V_DIM; c += 256) {
            bool skip = false;
            for (int j = 0; j < t; ++j) if (sel[j] == c) skip = true;
            if (skip) continue;
            float v = logits[(size_t)r * V_DIM + c];
            if (v > bv || (v == bv && c < bi)) { bv = v; bi = c; }
        }
        rv[tid] = bv; ri[tid] = bi; __syncthreads();
        for (int s = 128; s > 0; s >>= 1) {
            if (tid < s) {
                if (rv[tid + s] > rv[tid] || (rv[tid + s] == rv[tid] && ri[tid + s] < ri[tid])) {
                    rv[tid] = rv[tid + s]; ri[tid] = ri[tid + s];
                }
            }
            __syncthreads();
        }
        if (tid == 0) {
            sel[t] = ri[0];
            topk_i[r * TOPK + t] = ri[0];
            topk_p[r * TOPK + t] = rv[0] - row_lse[r];
        }
        __syncthreads();
    }
}

__global__ void k_init_state(const float* __restrict__ topk_p, const int* __restrict__ topk_i,
                             float* scores, int* topk_cs, int* cur_ids,
                             float* scores_list, int* tokens_list, int* parents_flat) {
    if (threadIdx.x != 0) return;
    for (int j = 0; j < TOPK; ++j) {
        scores[j]      = topk_p[j];
        scores_list[j] = topk_p[j];
        tokens_list[j] = topk_i[j];
        cur_ids[j]     = topk_i[j];
        topk_cs[j]     = j;
    }
    parents_flat[0] = 0;
}

__global__ void k_iter_state(int iter, const float* __restrict__ topk_p,
                             const int* __restrict__ topk_i,
                             float* scores, int* topk_cs, int* cur_ids, int* out_ids,
                             float* scores_list, int* tokens_list, int* parents_flat) {
    if (threadIdx.x != 0) return;
    int bias1 = (iter > 0) ? TOPK : 0;
    int bias2 = (iter > 1) ? (iter - 1) : 0;
    int bias  = 1 + TOPK * TOPK * bias2 + bias1;
    for (int j = 0; j < TOPK; ++j) parents_flat[1 + iter * TOPK + j] = topk_cs[j] + bias;

    float cu[TOPK * TOPK];
    for (int r = 0; r < TOPK; ++r)
        for (int c = 0; c < TOPK; ++c) {
            float v = topk_p[r * TOPK + c] + scores[r];
            cu[r * TOPK + c] = v;
            scores_list[TOPK + iter * 100 + r * TOPK + c] = v;
            tokens_list[TOPK + iter * 100 + r * TOPK + c] = topk_i[r * TOPK + c];
        }
    bool used[TOPK * TOPK] = {false};
    float ns[TOPK]; int ni[TOPK];
    for (int t = 0; t < TOPK; ++t) {
        float bv = -INFINITY; int bi = 0;
        for (int j = 0; j < TOPK * TOPK; ++j)
            if (!used[j] && cu[j] > bv) { bv = cu[j]; bi = j; }
        used[bi] = true; ns[t] = bv; ni[t] = bi;
    }
    for (int t = 0; t < TOPK; ++t) {
        scores[t]  = ns[t];
        topk_cs[t] = ni[t];
        out_ids[t] = ni[t] / TOPK;
        cur_ids[t] = topk_i[ni[t]];
    }
}

__global__ void k_finalize(const int* __restrict__ input_ids,
                           const float* __restrict__ scores_list,
                           const int* __restrict__ tokens_list,
                           const int* __restrict__ parents_flat,
                           float* __restrict__ out) {
    __shared__ int top_idx[TOTAL];
    __shared__ int mask_index[TOTAL];
    __shared__ unsigned long long maskrows[TOTAL + 1];
    int tid = threadIdx.x;
    const int NSC = TOPK + DEPTH * TOPK * TOPK; // 510
    if (tid == 0) {
        bool used[510] = {false};
        for (int t = 0; t < TOTAL; ++t) {
            float bv = -INFINITY; int bi = 0;
            for (int j = 0; j < NSC; ++j)
                if (!used[j] && scores_list[j] > bv) { bv = scores_list[j]; bi = j; }
            used[bi] = true;
        }
        int n = 0;
        for (int j = 0; j < NSC; ++j) if (used[j]) top_idx[n++] = j;

        out[0] = (float)input_ids[S_LEN];            // sample token
        for (int t = 0; t < TOTAL; ++t) {
            out[1 + t]  = (float)tokens_list[top_idx[t]];   // draft_tokens
            out[60 + t] = scores_list[top_idx[t]];          // top_vals
        }
        for (int t = 0; t < TOTAL; ++t) {
            int dp = parents_flat[top_idx[t] / TOPK];
            int mi;
            if (dp == 0) mi = -1;
            else {
                int target = dp - 1, lo = 0;
                while (lo < TOTAL && top_idx[lo] < target) lo++;  // searchsorted-left
                mi = lo;
            }
            mask_index[t] = mi + 1;
        }
        for (int r = 0; r < TOTAL + 1; ++r) maskrows[r] = (1ull << r) | 1ull;
        for (int t = 0; t < TOTAL; ++t) maskrows[t + 1] |= maskrows[mask_index[t]];
    }
    __syncthreads();
    for (int e = tid; e < (TOTAL + 1) * (TOTAL + 1); e += blockDim.x) {
        int r = e / (TOTAL + 1), c = e % (TOTAL + 1);
        out[119 + e] = ((maskrows[r] >> c) & 1ull) ? 1.f : 0.f;
    }
    if (tid < TOTAL + 1)
        out[119 + 3600 + tid] = (float)(__popcll(maskrows[tid]) - 1);
}

// ---------------------------------------------------------------------------
extern "C" void kernel_launch(void* const* d_in, const int* in_sizes, int n_in,
                              void* d_out, int out_size, void* d_ws, size_t ws_size,
                              hipStream_t stream) {
    const float* hs        = (const float*)d_in[0];
    const int*   input_ids = (const int*)d_in[1];
    const float* W_embed   = (const float*)d_in[2];
    const float* W_head    = (const float*)d_in[3];
    const float* W_fc      = (const float*)d_in[4];
    const float* b_fc      = (const float*)d_in[5];
    const float* W_m       = (const float*)d_in[6];
    const float* b_m       = (const float*)d_in[7];
    float* out = (float*)d_out;

    char* ws = (char*)d_ws;
    auto carve = [&](size_t bytes) { char* p = ws; ws += (bytes + 255) & ~(size_t)255; return p; };
    __bf16* abuf_fc   = (__bf16*)carve((size_t)MPAD * 2 * H_DIM * 2);  // [emb|hid] bf16
    __bf16* abuf_m    = (__bf16*)carve((size_t)MPAD * H_DIM * 2);      // h_fc bf16
    __bf16* abuf_head = (__bf16*)carve((size_t)MPAD * H_DIM * 2);      // out_hid bf16
    float*  out_hid   = (float*)carve((size_t)MPAD * H_DIM * 4);       // fp32 (for gathers)
    float*  logits    = (float*)carve((size_t)TOPK * V_DIM * 4);
    float*  row_lse   = (float*)carve(16 * 4);
    float*  topk_p    = (float*)carve(TOPK * TOPK * 4);
    int*    topk_i    = (int*)  carve(TOPK * TOPK * 4);
    float*  scores    = (float*)carve(TOPK * 4);
    int*    topk_cs   = (int*)  carve(TOPK * 4);
    int*    cur_ids   = (int*)  carve(16 * 4);
    int*    out_ids   = (int*)  carve(TOPK * 4);
    float*  scores_list  = (float*)carve(512 * 4);
    int*    tokens_list  = (int*)  carve(512 * 4);
    int*    parents_flat = (int*)  carve(64 * 4);

    dim3 gRow(H_DIM / 256, MPAD);
    int fc_blocks   = (H_DIM / 16) / 4;   // 64 blocks of 4 waves
    int head_blocks = (V_DIM / 16) / 4;   // 500 blocks of 4 waves

    // ---- initial draft step: only the last position matters ----
    k_embed<<<gRow, 256, 0, stream>>>(input_ids + S_LEN, W_embed, abuf_fc, 1);
    k_set_hidden<<<gRow, 256, 0, stream>>>(hs + (size_t)(S_LEN - 1) * H_DIM, nullptr, abuf_fc, 0, 1);
    k_gemm<<<fc_blocks, 128, 0, stream>>>(abuf_fc, W_fc, b_fc, nullptr, abuf_m,
                                          H_DIM, 2 * H_DIM, H_DIM, H_DIM, 1, MPAD);
    k_gemm<<<fc_blocks, 128, 0, stream>>>(abuf_m, W_m, b_m, out_hid, abuf_head,
                                          H_DIM, H_DIM, H_DIM, H_DIM, 2, MPAD);
    k_gemm<<<head_blocks, 128, 0, stream>>>(abuf_head, W_head, nullptr, logits, nullptr,
                                            V_DIM, H_DIM, V_DIM, 0, 0, TOPK);
    k_row_reduce<<<1, 256, 0, stream>>>(logits, row_lse);
    k_topk_row<<<1, 256, 0, stream>>>(logits, row_lse, topk_p, topk_i);
    k_init_state<<<1, 32, 0, stream>>>(topk_p, topk_i, scores, topk_cs, cur_ids,
                                       scores_list, tokens_list, parents_flat);
    // input_hidden = broadcast(last_hidden) over TOPK rows
    k_set_hidden<<<gRow, 256, 0, stream>>>(out_hid, nullptr, abuf_fc, 0, TOPK);

    // ---- DEPTH tree levels ----
    for (int i = 0; i < DEPTH; ++i) {
        k_embed<<<gRow, 256, 0, stream>>>(cur_ids, W_embed, abuf_fc, TOPK);
        k_gemm<<<fc_blocks, 128, 0, stream>>>(abuf_fc, W_fc, b_fc, nullptr, abuf_m,
                                              H_DIM, 2 * H_DIM, H_DIM, H_DIM, 1, MPAD);
        k_gemm<<<fc_blocks, 128, 0, stream>>>(abuf_m, W_m, b_m, out_hid, abuf_head,
                                              H_DIM, H_DIM, H_DIM, H_DIM, 2, MPAD);
        k_gemm<<<head_blocks, 128, 0, stream>>>(abuf_head, W_head, nullptr, logits, nullptr,
                                                V_DIM, H_DIM, V_DIM, 0, 0, TOPK);
        k_row_reduce<<<TOPK, 256, 0, stream>>>(logits, row_lse);
        k_topk_row<<<TOPK, 256, 0, stream>>>(logits, row_lse, topk_p, topk_i);
        k_iter_state<<<1, 32, 0, stream>>>(i, topk_p, topk_i, scores, topk_cs,
                                           cur_ids, out_ids,
                                           scores_list, tokens_list, parents_flat);
        if (i < DEPTH - 1)
            k_set_hidden<<<gRow, 256, 0, stream>>>(out_hid, out_ids, abuf_fc, 1, TOPK);
    }

    k_finalize<<<1, 64, 0, stream>>>(input_ids, scores_list, tokens_list, parents_flat, out);
}
